// StackedLSTMModel_3994319585457
// MI455X (gfx1250) — matrix operand order repaired and checked by reading, compiled, not run on hardware
//
#include <hip/hip_runtime.h>
#include <hip/hip_bf16.h>

// ---------------------------------------------------------------------------
// Stacked LSTM (B=2048, T=256, F=64 -> H 64/32/16 -> Dense 1) for gfx1250.
// Fused per-layer recurrent kernels: all GEMMs via v_wmma_f32_16x16x32_f16,
// weights persistent in VGPR B-fragments, h via LDS, c in registers.
// Gate nonlinearities on the TRANS unit (v_tanh_f32) to co-execute with WMMA.
// f16 layer inputs staged with global_load_async_to_lds_b128 (ASYNCcnt path).
// ---------------------------------------------------------------------------

typedef _Float16 h16x16 __attribute__((ext_vector_type(16)));
typedef _Float16 h16x8  __attribute__((ext_vector_type(8)));
typedef _Float16 h16x4  __attribute__((ext_vector_type(4)));
typedef float    f32x8  __attribute__((ext_vector_type(8)));
typedef float    f32x4  __attribute__((ext_vector_type(4)));

#if __has_builtin(__builtin_amdgcn_tanhf)
__device__ __forceinline__ float fast_tanh(float x) {
    return __builtin_amdgcn_tanhf(x);           // v_tanh_f32 (TRANS)
}
__device__ __forceinline__ float fast_sigmoid(float x) {
    return fmaf(0.5f, __builtin_amdgcn_tanhf(0.5f * x), 0.5f);
}
#else
__device__ __forceinline__ float fast_tanh(float x) {
    x = fminf(15.0f, fmaxf(-15.0f, x));
    float e = __expf(-2.0f * x);                 // v_exp_f32
    return (1.0f - e) * __builtin_amdgcn_rcpf(1.0f + e);  // v_rcp_f32
}
__device__ __forceinline__ float fast_sigmoid(float x) {
    return __builtin_amdgcn_rcpf(1.0f + __expf(-x));
}
#endif

// 32-bit LDS byte address of a shared-memory pointer (generic LDS pointers
// carry the DS offset in the low 32 bits; hardware truncates addr[31:0]).
__device__ __forceinline__ unsigned lds_addr_u32(const void* p) {
    return (unsigned)(uintptr_t)p;
}

// Build a 32x16 f16 B-fragment (K x N tile) from a row-major f32 weight
// [Krows x N4H], converting on the fly. Rows k >= Krows are zero (pads the
// H=16 recurrent K up to the WMMA K=32).
// Layout (16-bit B, wave32): lane = column n (lane&15); lanes 0-15: elements
// 0..7 = K kb..kb+7, 8..15 = K kb+16..kb+23; lanes 16-31: +8 on both chunks.
template<int N4H>
__device__ __forceinline__ h16x16 load_b_frag(const float* __restrict__ Wg,
                                              int Krows, int kb, int nb, int lane) {
    const int n  = nb + (lane & 15);
    const int kg = lane >> 4;
    h16x16 b;
#pragma unroll
    for (int e = 0; e < 16; ++e) {
        const int k = kb + kg * 8 + (e < 8 ? e : e + 8);
        float v = (k < Krows) ? Wg[(size_t)k * N4H + n] : 0.0f;
        b[e] = (_Float16)v;
    }
    return b;
}

// Load a 16x32 f16 A-fragment from an LDS tile (row stride stride_h halves).
// Layout (16-bit A, wave32): lane row m = lane&15; lanes 0-15: elements 0..7 =
// K kb..kb+7, elements 8..15 = K kb+16..kb+23; lanes 16-31: +8 on both chunks.
__device__ __forceinline__ h16x16 load_a_frag(const _Float16* buf, int stride_h,
                                              int row_base, int kb, int lane) {
    const int m = lane & 15;
    const int g = lane >> 4;
    const _Float16* p = buf + (size_t)(row_base + m) * stride_h + kb + 8 * g;
    h16x8 c0 = *(const h16x8*)p;          // ds_load_b128 (16B aligned)
    h16x8 c1 = *(const h16x8*)(p + 16);
    h16x16 a;
#pragma unroll
    for (int i = 0; i < 8; ++i) { a[i] = c0[i]; a[i + 8] = c1[i]; }
    return a;
}

// One LSTM layer, full scan over T. Grid: B / (16*NW_B) blocks of 32*NW_B*NW_H
// threads. Wave w handles batch subtile (w/NW_H) and hidden-column tile
// (w%NW_H). Gate order i,f,g,o along the 4H axis (Keras convention).
template<typename XT, int IN, int H, int NW_B, int NW_H, bool STORE_SEQ, bool LAST>
__global__ __launch_bounds__(32 * NW_B * NW_H)
void lstm_layer_kernel(const XT* __restrict__ xin,    // [B, T, IN]
                       const float* __restrict__ W,   // [IN, 4H] f32
                       const float* __restrict__ U,   // [H, 4H]  f32
                       const float* __restrict__ bias,// [4H]     f32
                       _Float16* __restrict__ hseq,   // [B, T, H] f16 out
                       float* __restrict__ hlast,     // [B, H]    f32 out
                       int T) {
    constexpr int NTH   = 32 * NW_B * NW_H;
    constexpr int BT    = 16 * NW_B;           // batch rows per block
    constexpr int KS_IN = IN / 32;             // WMMA k-steps over input dim
    constexpr int HK    = (H < 32) ? 32 : H;   // recurrent K padded to >=32
    constexpr int KS_H  = HK / 32;
    constexpr int SWX   = IN + 8;              // LDS row strides (halves)
    constexpr int SWH   = HK + 8;
    constexpr int N4H   = 4 * H;

    __shared__ __align__(16) _Float16 xbuf[BT * SWX];
    __shared__ __align__(16) _Float16 hbuf[BT * SWH];

    const int tid  = threadIdx.x;
    const int lane = tid & 31;
    const int w    = tid >> 5;
    const int bs   = w / NW_H;                 // batch subtile of this wave
    const int ht   = w % NW_H;                 // hidden-column tile
    const int b0   = blockIdx.x * BT;

    // h(t=-1) = 0 (including the zero-pad columns H..HK for layer H=16)
    for (int i = tid; i < BT * SWH; i += NTH) hbuf[i] = (_Float16)0.0f;

    // Loop-invariant weight fragments: held in VGPRs for all 256 timesteps.
    h16x16 BW[4][KS_IN];
    h16x16 BU[4][KS_H];
    float  bv[4];
#pragma unroll
    for (int gt = 0; gt < 4; ++gt) {
        const int nb = gt * H + ht * 16;
#pragma unroll
        for (int ks = 0; ks < KS_IN; ++ks)
            BW[gt][ks] = load_b_frag<N4H>(W, IN, 32 * ks, nb, lane);
#pragma unroll
        for (int ks = 0; ks < KS_H; ++ks)
            BU[gt][ks] = load_b_frag<N4H>(U, H, 32 * ks, nb, lane);
        bv[gt] = bias[nb + (lane & 15)];
    }

    f32x8 cst;                                  // cell state tile (16x16) f32
#pragma unroll
    for (int r = 0; r < 8; ++r) cst[r] = 0.0f;

    __syncthreads();

    for (int t = 0; t < T; ++t) {
        // ---- stage x(:, t, :) into LDS as f16 (coalesced, whole block) ----
        if constexpr (sizeof(XT) == 4) {        // f32 input (layer 0): convert
            constexpr int CPR = IN / 4;         // float4 chunks per row
            for (int q = tid; q < BT * CPR; q += NTH) {
                const int row = q / CPR, cq = q % CPR;
                const size_t base = (size_t)(b0 + row) * T * IN + (size_t)t * IN;
                const f32x4 v = *(const f32x4*)((const float*)xin + base + cq * 4);
                h16x4 hv;
#pragma unroll
                for (int i = 0; i < 4; ++i) hv[i] = (_Float16)v[i];
                *(h16x4*)(&xbuf[row * SWX + cq * 4]) = hv;
                if (t + 1 < T)
                    __builtin_prefetch((const float*)xin + base + IN + cq * 4, 0, 3);
            }
        } else {                                // f16 input: async copy to LDS
            constexpr int CPR = IN / 8;         // 16-byte chunks per row
            for (int q = tid; q < BT * CPR; q += NTH) {
                const int row = q / CPR, cq = q % CPR;
                const size_t base = (size_t)(b0 + row) * T * IN + (size_t)t * IN;
                const unsigned goff = (unsigned)((base + (size_t)cq * 8) * sizeof(_Float16));
                const unsigned ldsb = lds_addr_u32(&xbuf[row * SWX + cq * 8]);
                // GVS mode: SGPR 64-bit base + per-lane 32-bit byte offset.
                asm volatile("global_load_async_to_lds_b128 %0, %1, %2"
                             :: "v"(ldsb), "v"(goff), "s"((const void*)xin)
                             : "memory");
                if (t + 1 < T)
                    __builtin_prefetch((const _Float16*)xin + base + IN + cq * 8, 0, 3);
            }
            asm volatile("s_wait_asynccnt 0" ::: "memory");
        }
        __syncthreads();   // xbuf ready; hbuf writes from t-1 visible

        // ---- A fragments (x part is h-independent: fills recurrence shadow)
        h16x16 AX[KS_IN], AH[KS_H];
#pragma unroll
        for (int ks = 0; ks < KS_IN; ++ks)
            AX[ks] = load_a_frag(xbuf, SWX, bs * 16, 32 * ks, lane);
#pragma unroll
        for (int ks = 0; ks < KS_H; ++ks)
            AH[ks] = load_a_frag(hbuf, SWH, bs * 16, 32 * ks, lane);

        // ---- z = bias + x@W + h@U via WMMA; ks outer so 4 consecutive WMMAs
        // share one A operand and form 4 independent accumulation chains.
        f32x8 acc[4];
#pragma unroll
        for (int gt = 0; gt < 4; ++gt)
#pragma unroll
            for (int r = 0; r < 8; ++r) acc[gt][r] = bv[gt];
#pragma unroll
        for (int ks = 0; ks < KS_IN; ++ks)
#pragma unroll
            for (int gt = 0; gt < 4; ++gt)
                acc[gt] = __builtin_amdgcn_wmma_f32_16x16x32_f16(
                    false, AX[ks], false, BW[gt][ks], (short)0, acc[gt], false, false);
#pragma unroll
        for (int ks = 0; ks < KS_H; ++ks)
#pragma unroll
            for (int gt = 0; gt < 4; ++gt)
                acc[gt] = __builtin_amdgcn_wmma_f32_16x16x32_f16(
                    false, AH[ks], false, BU[gt][ks], (short)0, acc[gt], false, false);

        __syncthreads();   // all reads of xbuf/hbuf done before we overwrite h

        // ---- gates (TRANS unit), state update, h writeback ----
        // C/D layout: lane l, reg r -> m = r + 8*(l>>4), n = l&15
        const int n = lane & 15;
        const int g = lane >> 4;
#pragma unroll
        for (int r = 0; r < 8; ++r) {
            const float iv = fast_sigmoid(acc[0][r]);
            const float fv = fast_sigmoid(acc[1][r]);
            const float gv = fast_tanh(acc[2][r]);
            const float ov = fast_sigmoid(acc[3][r]);
            const float cv = fmaf(fv, cst[r], iv * gv);
            cst[r] = cv;
            const float hv = ov * fast_tanh(cv);
            const int row = bs * 16 + r + 8 * g;
            hbuf[row * SWH + ht * 16 + n] = (_Float16)hv;
            if constexpr (STORE_SEQ)
                hseq[((size_t)(b0 + row) * T + t) * H + ht * 16 + n] = (_Float16)hv;
            if constexpr (LAST)
                if (t == T - 1)
                    hlast[(size_t)(b0 + row) * H + ht * 16 + n] = hv;
        }
    }
}

// Final Dense(16 -> 1): out[b] = h_last[b,:] @ Wd + bd
__global__ void dense_kernel(const float* __restrict__ hlast,
                             const float* __restrict__ Wd,
                             const float* __restrict__ bd,
                             float* __restrict__ out, int Bsz) {
    const int i = blockIdx.x * blockDim.x + threadIdx.x;
    if (i < Bsz) {
        float s = bd[0];
#pragma unroll
        for (int j = 0; j < 16; ++j) s += hlast[(size_t)i * 16 + j] * Wd[j];
        out[i] = s;
    }
}

extern "C" void kernel_launch(void* const* d_in, const int* in_sizes, int n_in,
                              void* d_out, int out_size, void* d_ws, size_t ws_size,
                              hipStream_t stream) {
    constexpr int B = 2048, T = 256;

    const float* x  = (const float*)d_in[0];
    const float* W0 = (const float*)d_in[1];
    const float* U0 = (const float*)d_in[2];
    const float* b0 = (const float*)d_in[3];
    const float* W1 = (const float*)d_in[4];
    const float* U1 = (const float*)d_in[5];
    const float* b1 = (const float*)d_in[6];
    const float* W2 = (const float*)d_in[7];
    const float* U2 = (const float*)d_in[8];
    const float* b2 = (const float*)d_in[9];
    const float* Wd = (const float*)d_in[10];
    const float* bd = (const float*)d_in[11];
    float* out = (float*)d_out;

    // Workspace: h0seq f16 [B,T,64] | h1seq f16 [B,T,32] | hlast f32 [B,16]
    char* ws = (char*)d_ws;
    _Float16* h0 = (_Float16*)ws;
    _Float16* h1 = (_Float16*)(ws + (size_t)B * T * 64 * sizeof(_Float16));
    float*    hl = (float*)(ws + (size_t)B * T * 64 * sizeof(_Float16)
                               + (size_t)B * T * 32 * sizeof(_Float16));

    dim3 blk(256);
    // Layer 0: IN=64, H=64. 8 waves = 2 batch subtiles x 4 hidden tiles.
    lstm_layer_kernel<float, 64, 64, 2, 4, true, false>
        <<<B / 32, blk, 0, stream>>>(x, W0, U0, b0, h0, nullptr, T);
    // Layer 1: IN=64, H=32. 4 batch subtiles x 2 hidden tiles.
    lstm_layer_kernel<_Float16, 64, 32, 4, 2, true, false>
        <<<B / 64, blk, 0, stream>>>(h0, W1, U1, b1, h1, nullptr, T);
    // Layer 2: IN=32, H=16 (recurrent K zero-padded to 32). 8 batch subtiles.
    lstm_layer_kernel<_Float16, 32, 16, 8, 1, false, true>
        <<<B / 128, blk, 0, stream>>>(h1, W2, U2, b2, nullptr, hl, T);
    // Dense 16 -> 1
    dense_kernel<<<B / 256, 256, 0, stream>>>(hl, Wd, bd, out, B);
}
